// GlobalMultiHeadAttention_23854248362060
// MI455X (gfx1250) — matrix operand order repaired and checked by reading, not compiled, not run on hardware
//
#include <hip/hip_runtime.h>

typedef __attribute__((ext_vector_type(16))) __bf16 v16bf;
typedef __attribute__((ext_vector_type(8)))  __bf16 v8bf;
typedef __attribute__((ext_vector_type(2)))  __bf16 v2bf;
typedef __attribute__((ext_vector_type(8)))  float  v8f;

__device__ __forceinline__ v16bf frag2(const __bf16* p0, const __bf16* p1) {
    v8bf lo = *(const v8bf*)p0;
    v8bf hi = *(const v8bf*)p1;
    v16bf r;
    #pragma unroll
    for (int e = 0; e < 8; ++e) { r[e] = lo[e]; r[8 + e] = hi[e]; }
    return r;
}

__device__ __forceinline__ void store_pk(__bf16* dst, float a, float b) {
    v2bf p; p[0] = (__bf16)a; p[1] = (__bf16)b;   // v_cvt_pk_bf16_f32
    *(v2bf*)dst = p;                               // ds_store_b32
}

// ---------------------------------------------------------------------------
// M=128 GEMM:  C[128, NTILES*16*gridX] = alpha*(X[128,K] @ W) (+bias)(+resid)(+=C)
// W: [K,N] row-major (WTRANS=0) or [N,K] row-major (WTRANS=1, i.e. B^T).
// grid.x tiles N in blocks of NTILES*16 cols; grid.y batches heads via strides.
// bf16 WMMA, f32 accumulation; paired bf16 staging (b64 load / b32 DS store).
// ---------------------------------------------------------------------------
template <int NTILES, bool WTRANS, bool BIAS, bool RESID, bool ACCUM>
__global__ __launch_bounds__(256)
void gemm_wmma(const float* __restrict__ X, int ldx,
               const float* __restrict__ W, int ldw,
               float* __restrict__ C, int ldc,
               int Kdim, float alpha,
               const float* __restrict__ bias,
               const float* __restrict__ resid,
               long long sxY, long long swY, long long scY)
{
    constexpr int NCOL  = NTILES * 16;   // columns per block
    constexpr int STR   = 40;            // padded LDS stride (80B, 16B-aligned)

    X += (long long)blockIdx.y * sxY;
    W += (long long)blockIdx.y * swY;
    C += (long long)blockIdx.y * scY;

    __shared__ alignas(16) __bf16 sA [128 * STR];  // [row][kk]
    __shared__ alignas(16) __bf16 sBt[128 * STR];  // [col][kk]  (transposed)

    const int tid  = threadIdx.x;
    const int lane = tid & 31;
    const int wave = tid >> 5;                     // 0..7, 16 rows each
    const int colBase = blockIdx.x * NCOL;

    v8f acc[NTILES];
    const v8f vzero = {0.f,0.f,0.f,0.f,0.f,0.f,0.f,0.f};
    #pragma unroll
    for (int t = 0; t < NTILES; ++t) acc[t] = vzero;

    const int rowA    = wave * 16 + (lane & 15);
    const int khalfA  = (lane >> 4) * 8;           // 0 or 8
    const int khalfB  = (lane >> 4) * 8;           // 0 or 8
    const int colLane = lane & 15;

    const __bf16* aP0 = sA + rowA * STR + khalfA;       // 8 elems
    const __bf16* aP1 = sA + rowA * STR + 16 + khalfA;  // 8 elems

    for (int kb = 0; kb < Kdim; kb += 32) {
        // ---- stage A tile (128 x 32): float2 load, packed bf16x2 DS store
        #pragma unroll
        for (int it = 0; it < 8; ++it) {
            int idx = tid + it * 256;              // pair index 0..2047
            int r  = idx >> 4;                     // 16 k-pairs per row
            int kk = (idx & 15) * 2;
            float2 v = *(const float2*)(X + r * ldx + kb + kk);
            store_pk(&sA[r * STR + kk], v.x, v.y);
        }
        // ---- stage B tile (32 x NCOL) into transposed layout
        if (WTRANS) {
            // memory contiguous along kk -> float2 along kk
            #pragma unroll
            for (int it = 0; it < NCOL / 16; ++it) {
                int idx = tid + it * 256;          // pair index 0 .. NCOL*16-1
                int c  = idx >> 4;
                int kk = (idx & 15) * 2;
                float2 v = *(const float2*)(W + (long long)(colBase + c) * ldw + kb + kk);
                store_pk(&sBt[c * STR + kk], v.x, v.y);
            }
        } else {
            // memory contiguous along columns -> pair across two W rows
            #pragma unroll
            for (int it = 0; it < NCOL / 16; ++it) {
                int idx = tid + it * 256;          // pair index 0 .. NCOL*16-1
                int c  = idx % NCOL;
                int kk = (idx / NCOL) * 2;
                float v0 = W[(long long)(kb + kk)     * ldw + colBase + c];
                float v1 = W[(long long)(kb + kk + 1) * ldw + colBase + c];
                store_pk(&sBt[c * STR + kk], v0, v1);
            }
        }
        __syncthreads();

        // ---- fragments: two 16B ds_load_b128 each
        v16bf afrag = frag2(aP0, aP1);
        #pragma unroll
        for (int t = 0; t < NTILES; ++t) {
            const __bf16* bP = sBt + (t * 16 + colLane) * STR + khalfB;
            v16bf bfrag = frag2(bP, bP + 16);
            acc[t] = __builtin_amdgcn_wmma_f32_16x16x32_bf16(
                false, afrag, false, bfrag, (short)0, acc[t], false, false);
        }
        __syncthreads();
    }

    // ---- store (D layout -> row-major C)
    #pragma unroll
    for (int t = 0; t < NTILES; ++t) {
        int col = colBase + t * 16 + colLane;
        #pragma unroll
        for (int v = 0; v < 8; ++v) {
            int row = wave * 16 + ((lane < 16) ? v : v + 8);
            float val = alpha * acc[t][v];
            if (BIAS)  val += bias[col];
            if (RESID) val += resid[row * ldc + col];
            if (ACCUM) val += C[row * ldc + col];
            C[row * ldc + col] = val;
        }
    }
}

// ---------------------------------------------------------------------------
// Decision mask. dec[p=i*3+j][q][k] = -1e9 if
//   sum_c relu(A_j[q,c] + B_i[k,c]) * (w2[c,1]-w2[c,0]) + (b2[1]-b2[0]) > 0
// (gumbel-softmax hard eval == argmax of logits; tie -> class 0 -> mask 0)
// ---------------------------------------------------------------------------
__global__ __launch_bounds__(128)
void dec_kernel(const float* __restrict__ Abuf,  // [3][128][2048] (b1 folded)
                const float* __restrict__ Bbuf,  // [3][128][2048]
                const float* __restrict__ w2,    // [2048][2]
                const float* __restrict__ b2,    // [2]
                float* __restrict__ decout)      // [9][128][128]
{
    __shared__ alignas(16) float sW2d[2048];
    __shared__ alignas(16) float sArow[2048];
    const int p = blockIdx.x, i = p / 3, j = p % 3;
    const int q = blockIdx.y;
    const int k = threadIdx.x;

    for (int c = threadIdx.x; c < 2048; c += 128) {
        sW2d[c]  = w2[c * 2 + 1] - w2[c * 2 + 0];
        sArow[c] = Abuf[(j * 128 + q) * 2048 + c];
    }
    __syncthreads();

    const float* Brow = Bbuf + (long long)(i * 128 + k) * 2048;
    float acc = b2[1] - b2[0];
    #pragma unroll 4
    for (int c = 0; c < 2048; c += 4) {
        float4 bv = *(const float4*)(Brow + c);
        float4 av = *(const float4*)(sArow + c);
        float4 wv = *(const float4*)(sW2d + c);
        float h0 = av.x + bv.x, h1 = av.y + bv.y, h2 = av.z + bv.z, h3 = av.w + bv.w;
        acc += (h0 > 0.f ? h0 : 0.f) * wv.x;
        acc += (h1 > 0.f ? h1 : 0.f) * wv.y;
        acc += (h2 > 0.f ? h2 : 0.f) * wv.z;
        acc += (h3 > 0.f ? h3 : 0.f) * wv.w;
    }
    decout[(p * 128 + q) * 128 + k] = (acc > 0.f) ? -1e9f : 0.f;
}

// ---------------------------------------------------------------------------
// In-place softmax over 384 keys of S[i][h][q][:], adding dec[i*3+j][q][k].
// grid = (q=128, h=16, i=3), block = 128.
// ---------------------------------------------------------------------------
__global__ __launch_bounds__(128)
void softmax_kernel(float* __restrict__ S,            // [3][16][128][384]
                    const float* __restrict__ decin)  // [9][128][128]
{
    const int q = blockIdx.x, h = blockIdx.y, i = blockIdx.z;
    const int tid = threadIdx.x;
    float* row = S + ((long long)((i * 16 + h) * 128) + q) * 384;

    float vals[3];
    float m = -3.4e38f;
    #pragma unroll
    for (int j = 0; j < 3; ++j) {
        float v = row[j * 128 + tid] + decin[((i * 3 + j) * 128 + q) * 128 + tid];
        vals[j] = v;
        m = fmaxf(m, v);
    }
    __shared__ float red[128];
    red[tid] = m; __syncthreads();
    for (int s = 64; s > 0; s >>= 1) {
        if (tid < s) red[tid] = fmaxf(red[tid], red[tid + s]);
        __syncthreads();
    }
    m = red[0]; __syncthreads();

    float sum = 0.f;
    #pragma unroll
    for (int j = 0; j < 3; ++j) { vals[j] = __expf(vals[j] - m); sum += vals[j]; }
    red[tid] = sum; __syncthreads();
    for (int s = 64; s > 0; s >>= 1) {
        if (tid < s) red[tid] += red[tid + s];
        __syncthreads();
    }
    const float inv = 1.f / red[0];
    #pragma unroll
    for (int j = 0; j < 3; ++j) row[j * 128 + tid] = vals[j] * inv;
}

// ---------------------------------------------------------------------------
// Row LayerNorm of F[i][q][:1024] -> out. grid=(q=128, i=3), block=256.
// One float4 per thread per pass.
// ---------------------------------------------------------------------------
__global__ __launch_bounds__(256)
void ln_kernel(const float* __restrict__ F,
               const float* __restrict__ g, const float* __restrict__ b,
               float* __restrict__ out)
{
    const int q = blockIdx.x, i = blockIdx.y;
    const float* row = F + (long long)(i * 128 + q) * 1024;
    float* orow = out + (long long)(i * 128 + q) * 1024;
    const int c0 = threadIdx.x * 4;
    const int tid = threadIdx.x;

    float4 v = *(const float4*)(row + c0);
    float s  = v.x + v.y + v.z + v.w;
    float s2 = v.x*v.x + v.y*v.y + v.z*v.z + v.w*v.w;

    __shared__ float rs[256], rs2[256];
    rs[tid] = s; rs2[tid] = s2; __syncthreads();
    for (int t = 128; t > 0; t >>= 1) {
        if (tid < t) { rs[tid] += rs[tid + t]; rs2[tid] += rs2[tid + t]; }
        __syncthreads();
    }
    const float mu  = rs[0] * (1.f / 1024.f);
    const float var = rs2[0] * (1.f / 1024.f) - mu * mu;
    const float inv = rsqrtf(var + 1e-6f);

    float4 gv = *(const float4*)(g + c0);
    float4 bv = *(const float4*)(b + c0);
    float4 o;
    o.x = (v.x - mu) * inv * gv.x + bv.x;
    o.y = (v.y - mu) * inv * gv.y + bv.y;
    o.z = (v.z - mu) * inv * gv.z + bv.z;
    o.w = (v.w - mu) * inv * gv.w + bv.w;
    *(float4*)(orow + c0) = o;
}

// ---------------------------------------------------------------------------
extern "C" void kernel_launch(void* const* d_in, const int* in_sizes, int n_in,
                              void* d_out, int out_size, void* d_ws, size_t ws_size,
                              hipStream_t stream) {
    const float* x[3]   = {(const float*)d_in[0], (const float*)d_in[1], (const float*)d_in[2]};
    const float* w_qs   = (const float*)d_in[3];
    const float* w_ks   = (const float*)d_in[4];
    const float* w_vs   = (const float*)d_in[5];
    const float* w_fc   = (const float*)d_in[6];
    const float* w_ds0  = (const float*)d_in[7];
    const float* w_ds1  = (const float*)d_in[8];
    const float* dc_w1  = (const float*)d_in[9];   // [2048][2048]
    const float* dc_b1  = (const float*)d_in[10];  // [2048]
    const float* dc_w2  = (const float*)d_in[11];  // [2048][2]
    const float* dc_b2  = (const float*)d_in[12];  // [2]
    const float* ln_g   = (const float*)d_in[13];
    const float* ln_b   = (const float*)d_in[14];

    float* ws = (float*)d_ws;
    float* Q   = ws;              // 3*128*1024
    float* Kb  = Q   + 393216;
    float* V   = Kb  + 393216;
    float* D0  = V   + 393216;
    float* D1  = D0  + 393216;
    float* A   = D1  + 393216;    // 3*128*2048, b1 folded in
    float* Bm  = A   + 786432;    // 3*128*2048
    float* dcb = Bm  + 786432;    // 9*128*128
    float* S   = dcb + 147456;    // 3*16*128*384 (reused in place for P)
    float* O   = S   + 2359296;   // 3*128*1024
    float* F   = O   + 393216;    // 3*128*1024

    // Pass 1: projections (K=1024, N=1024)
    for (int i = 0; i < 3; ++i) {
        gemm_wmma<8,false,false,false,false><<<dim3(8,1,1),256,0,stream>>>(x[i],1024, w_qs ,1024, Q +i*131072,1024, 1024,1.f, nullptr,nullptr, 0,0,0);
        gemm_wmma<8,false,false,false,false><<<dim3(8,1,1),256,0,stream>>>(x[i],1024, w_ks ,1024, Kb+i*131072,1024, 1024,1.f, nullptr,nullptr, 0,0,0);
        gemm_wmma<8,false,false,false,false><<<dim3(8,1,1),256,0,stream>>>(x[i],1024, w_vs ,1024, V +i*131072,1024, 1024,1.f, nullptr,nullptr, 0,0,0);
        gemm_wmma<8,false,false,false,false><<<dim3(8,1,1),256,0,stream>>>(x[i],1024, w_ds0,1024, D0+i*131072,1024, 1024,1.f, nullptr,nullptr, 0,0,0);
        gemm_wmma<8,false,false,false,false><<<dim3(8,1,1),256,0,stream>>>(x[i],1024, w_ds1,1024, D1+i*131072,1024, 1024,1.f, nullptr,nullptr, 0,0,0);
    }
    // Pass 2: decision-MLP first layer, split by concat halves.
    // A_j = D1_j @ W1[:1024,:] + b1 ;  B_i = D0_i @ W1[1024:,:]
    for (int j = 0; j < 3; ++j)
        gemm_wmma<8,false,true,false,false><<<dim3(16,1,1),256,0,stream>>>(D1+j*131072,1024, dc_w1,2048,
            A+j*262144,2048, 1024,1.f, dc_b1,nullptr, 0,0,0);
    for (int i = 0; i < 3; ++i)
        gemm_wmma<8,false,false,false,false><<<dim3(16,1,1),256,0,stream>>>(D0+i*131072,1024, dc_w1+(long long)1024*2048,2048,
            Bm+i*262144,2048, 1024,1.f, nullptr,nullptr, 0,0,0);

    // Decision masks (9 pairs x 128 q rows)
    dec_kernel<<<dim3(9,128),128,0,stream>>>(A, Bm, dc_w2, dc_b2, dcb);

    // Scores: S[i][h][:, j*128:(j+1)*128] = (Q_i,h @ K_j,h^T) / 8, batched over h
    for (int i = 0; i < 3; ++i)
        for (int j = 0; j < 3; ++j)
            gemm_wmma<8,true,false,false,false><<<dim3(1,16,1),256,0,stream>>>(Q+i*131072,1024, Kb+j*131072,1024,
                S + (long long)i*16*49152 + j*128, 384, 64,0.125f, nullptr,nullptr, 64,64,49152);

    // Softmax over 384 keys with decision mask (in place -> P)
    softmax_kernel<<<dim3(128,16,3),128,0,stream>>>(S, dcb);

    // O_i[:, h*64:] = sum_j P[i][h][:, j*128:] @ V_j[:, h*64:], batched over h
    for (int i = 0; i < 3; ++i) {
        gemm_wmma<4,false,false,false,false><<<dim3(1,16,1),256,0,stream>>>(S + (long long)i*16*49152 + 0*128, 384,
            V+0*131072,1024, O+i*131072,1024, 128,1.f, nullptr,nullptr, 49152,64,64);
        gemm_wmma<4,false,false,false,true ><<<dim3(1,16,1),256,0,stream>>>(S + (long long)i*16*49152 + 1*128, 384,
            V+1*131072,1024, O+i*131072,1024, 128,1.f, nullptr,nullptr, 49152,64,64);
        gemm_wmma<4,false,false,false,true ><<<dim3(1,16,1),256,0,stream>>>(S + (long long)i*16*49152 + 2*128, 384,
            V+2*131072,1024, O+i*131072,1024, 128,1.f, nullptr,nullptr, 49152,64,64);
    }

    // fc + residual
    for (int i = 0; i < 3; ++i)
        gemm_wmma<8,false,false,true,false><<<dim3(8,1,1),256,0,stream>>>(O+i*131072,1024, w_fc,1024,
            F+i*131072,1024, 1024,1.f, nullptr,x[i], 0,0,0);

    // LayerNorm -> output [3,1,128,1024]
    ln_kernel<<<dim3(128,3),256,0,stream>>>(F, ln_g, ln_b, (float*)d_out);
}
